// Block_52527450030210
// MI455X (gfx1250) — compile-verified, matrix operand order checked by reading, not run on hardware
//
#include <hip/hip_runtime.h>
#include <hip/hip_bf16.h>
#include <math.h>

typedef __bf16 v16bf __attribute__((ext_vector_type(16)));
typedef float  v8f   __attribute__((ext_vector_type(8)));
typedef unsigned int v4u __attribute__((ext_vector_type(4)));

#define Bq 8
#define Sq 1024
#define Eq 1024
#define Hq 16
#define Fq 4096
#define Dq 64
#define BSq (Bq*Sq)

__device__ __forceinline__ unsigned short f2bf(float f){
  unsigned int u = __float_as_uint(f);
  u += 0x7FFFu + ((u >> 16) & 1u);
  return (unsigned short)(u >> 16);
}
__device__ __forceinline__ float bf2f(unsigned short h){
  return __uint_as_float(((unsigned int)h) << 16);
}

// CDNA5 async global->LDS 16B copy (ASYNCcnt-tracked, no VGPR round trip).
__device__ __forceinline__ void async_copy16(const void* g, void* l){
  unsigned int laddr = (unsigned int)(unsigned long long)l;  // LDS aperture: low 32 bits = LDS offset
  asm volatile("global_load_async_to_lds_b128 %0, %1, off"
               :: "v"(laddr), "v"((unsigned long long)g) : "memory");
}
__device__ __forceinline__ void wait_async_le3(){ asm volatile("s_wait_asynccnt 0x3" ::: "memory"); }
__device__ __forceinline__ void wait_async_le0(){ asm volatile("s_wait_asynccnt 0x0" ::: "memory"); }

__global__ void cvt_bf16_kernel(const float* __restrict__ in, unsigned short* __restrict__ out, int n){
  int i = blockIdx.x * blockDim.x + threadIdx.x;
  int stride = gridDim.x * blockDim.x;
  for (; i < n; i += stride) out[i] = f2bf(in[i]);
}

// fp32 [R,C] -> bf16 [C,R]  (coalesced both sides via LDS tile)
__global__ __launch_bounds__(256) void tcvt_kernel(const float* __restrict__ in,
                                                   unsigned short* __restrict__ out,
                                                   int R, int C){
  __shared__ float t[32][33];
  const int r0 = blockIdx.y * 32, c0 = blockIdx.x * 32;
  const int tx = threadIdx.x & 31, ty = threadIdx.x >> 5;
  #pragma unroll
  for (int i = ty; i < 32; i += 8) t[i][tx] = in[(long long)(r0 + i) * C + c0 + tx];
  __syncthreads();
  #pragma unroll
  for (int i = ty; i < 32; i += 8) out[(long long)(c0 + i) * R + r0 + tx] = f2bf(t[tx][i]);
}

// Vb bf16 [B*S, E] -> Vt bf16 [B*H, D, S]
__global__ __launch_bounds__(256) void vtrans_kernel(const unsigned short* __restrict__ Vb,
                                                     unsigned short* __restrict__ Vt){
  __shared__ unsigned short t[32][33];
  const int z = blockIdx.z;
  const int s0 = blockIdx.x * 32, d0 = blockIdx.y * 32;
  const int tx = threadIdx.x & 31, ty = threadIdx.x >> 5;
  const unsigned short* src = Vb + (long long)(z >> 4) * Sq * Eq + (z & 15) * Dq;
  #pragma unroll
  for (int i = ty; i < 32; i += 8) t[i][tx] = src[(long long)(s0 + i) * Eq + d0 + tx];
  __syncthreads();
  unsigned short* dst = Vt + (long long)z * Dq * Sq;
  #pragma unroll
  for (int i = ty; i < 32; i += 8) dst[(long long)(d0 + i) * Sq + s0 + tx] = t[tx][i];
}

// Z[z,k] = sum_q e[z,q,k]   (e is bf16, coalesced across 256 columns per block)
__global__ __launch_bounds__(256) void colsum_kernel(const unsigned short* __restrict__ e,
                                                     float* __restrict__ Z){
  const long long z = blockIdx.y;
  const int col = blockIdx.x * 256 + threadIdx.x;
  const unsigned short* sp = e + z * (long long)(Sq * Sq) + col;
  float s = 0.0f;
  for (int q = 0; q < Sq; ++q) s += bf2f(sp[(long long)q * Sq]);
  Z[z * Sq + col] = s;
}

// Vs[z,d,s] = Vt[z,d,s] / Z[z,s]   (folds the softmax normalizer into V)
__global__ __launch_bounds__(256) void vscale_kernel(const unsigned short* __restrict__ Vt,
                                                     const float* __restrict__ Z,
                                                     unsigned short* __restrict__ Vs){
  const int z = blockIdx.y, d = blockIdx.x;
  const unsigned short* src = Vt + ((long long)z * Dq + d) * Sq;
  unsigned short* dst = Vs + ((long long)z * Dq + d) * Sq;
  const float* Zp = Z + (long long)z * Sq;
  for (int s = threadIdx.x; s < Sq; s += 256)
    dst[s] = f2bf(bf2f(src[s]) / Zp[s]);
}

union Frag { v16bf v; v4u q[2]; };

// C[M,N] = act( scale * A_bf16[M,K] @ B_bf16[K,N] + bias )
// A row-major [M,K]; B source is [N,K] row-major (pre-transposed).
// Batch z: operand element offset = (z/HH)*sb + (z%HH)*sh.
// act: 0 = none, 1 = exact GELU, 2 = exp (for unnormalized softmax numerator)
__global__ __launch_bounds__(256) void gemm_bf16_wmma(
    const unsigned short* __restrict__ A, int lda, long long a_sb, long long a_sh,
    const unsigned short* __restrict__ Bm, int ldb, long long b_sb, long long b_sh,
    const float* __restrict__ bias,
    void* __restrict__ Cv, int ldc, long long c_sb, long long c_sh, int c_bf16,
    int M, int N, int K, int HH, float scale, int act)
{
  // double-buffered tiles: [row/col][16 k-pairs], padded to 20 (bank-conflict-free, 16B aligned)
  __shared__ __attribute__((aligned(16))) unsigned int lds_a[2][128][20];
  __shared__ __attribute__((aligned(16))) unsigned int lds_b[2][64][20];

  const int tid  = threadIdx.x;
  const int lane = tid & 31;
  const int w    = tid >> 5;
  const int wm   = w & 3;      // 0..3 -> 32-row slice
  const int wn   = w >> 2;     // 0..1 -> 32-col slice
  const int z    = blockIdx.z;

  const long long aoff = (long long)(z / HH) * a_sb + (long long)(z % HH) * a_sh;
  const long long boff = (long long)(z / HH) * b_sb + (long long)(z % HH) * b_sh;
  const long long coff = (long long)(z / HH) * c_sb + (long long)(z % HH) * c_sh;

  const int bm = blockIdx.y * 128;
  const int bn = blockIdx.x * 64;

  const unsigned short* Ag = A + aoff;
  const unsigned short* Bg = Bm + boff;

  v8f c[2][2] = {};

  const int lr  = lane & 15;
  const int kb2 = (lane >> 4) * 4;

  // issue one 128x32 A tile + 64x32 B tile as async 16B chunks (3 per thread)
  auto issue_tile = [&](int k0, int buf){
    #pragma unroll
    for (int t = 0; t < 2; ++t){
      int idx = tid + t * 256;          // 512 chunks
      int row = idx >> 2, j = idx & 3;
      async_copy16(Ag + (long long)(bm + row) * lda + k0 + 8 * j, &lds_a[buf][row][4 * j]);
    }
    {
      int row = tid >> 2, j = tid & 3;  // 256 chunks
      async_copy16(Bg + (long long)(bn + row) * ldb + k0 + 8 * j, &lds_b[buf][row][4 * j]);
    }
  };

  const int nt = K >> 5;
  issue_tile(0, 0);

  for (int it = 0; it < nt; ++it){
    const int cur = it & 1;
    const bool pf = (it + 1) < nt;
    if (pf) issue_tile((it + 1) << 5, cur ^ 1);   // prefetch next tile into other buffer
    if (pf) wait_async_le3(); else wait_async_le0();
    __syncthreads();

    Frag af[2], bfr[2];
    #pragma unroll
    for (int i = 0; i < 2; ++i){
      int r = wm * 32 + i * 16 + lr;
      af[i].q[0] = *(const v4u*)&lds_a[cur][r][kb2];
      af[i].q[1] = *(const v4u*)&lds_a[cur][r][8 + kb2];
    }
    #pragma unroll
    for (int j = 0; j < 2; ++j){
      int n = wn * 32 + j * 16 + lr;
      bfr[j].q[0] = *(const v4u*)&lds_b[cur][n][kb2];
      bfr[j].q[1] = *(const v4u*)&lds_b[cur][n][8 + kb2];
    }
    #pragma unroll
    for (int i = 0; i < 2; ++i)
      #pragma unroll
      for (int j = 0; j < 2; ++j)
        c[i][j] = __builtin_amdgcn_wmma_f32_16x16x32_bf16(
            false, af[i].v, false, bfr[j].v, (short)0, c[i][j], false, false);
    __syncthreads();   // all waves done reading buf `cur` before it is refilled at it+2
  }

  const int mhalf = (lane < 16) ? 0 : 8;
  float* Cf = (float*)Cv;
  unsigned short* Cb = (unsigned short*)Cv;
  #pragma unroll
  for (int i = 0; i < 2; ++i){
    #pragma unroll
    for (int j = 0; j < 2; ++j){
      int nn = bn + wn * 32 + j * 16 + lr;
      float bb = bias ? bias[nn] : 0.0f;
      #pragma unroll
      for (int r = 0; r < 8; ++r){
        int mm = bm + wm * 32 + i * 16 + mhalf + r;
        if (mm >= M || nn >= N) continue;
        float val = c[i][j][r] * scale + bb;
        if (act == 1) val = 0.5f * val * (1.0f + erff(val * 0.70710678118f));
        else if (act == 2) val = __expf(val);
        long long o = coff + (long long)mm * ldc + nn;
        if (c_bf16) Cb[o] = f2bf(val); else Cf[o] = val;
      }
    }
  }
}

// adn1 = x + LayerNorm(att_out)*g + b
__global__ __launch_bounds__(256) void addnorm1_kernel(
    const float* __restrict__ x, const float* __restrict__ ao,
    const float* __restrict__ g, const float* __restrict__ b,
    float* __restrict__ adn1, unsigned short* __restrict__ adn1b){
  __shared__ float rs_[256], rq_[256];
  const int row = blockIdx.x, tid = threadIdx.x;
  const float* ar = ao + (long long)row * Eq;
  float v[4]; float s = 0.f, sq = 0.f;
  #pragma unroll
  for (int j = 0; j < 4; ++j){ v[j] = ar[tid + j*256]; s += v[j]; sq += v[j]*v[j]; }
  rs_[tid] = s; rq_[tid] = sq; __syncthreads();
  for (int off = 128; off > 0; off >>= 1){
    if (tid < off){ rs_[tid] += rs_[tid+off]; rq_[tid] += rq_[tid+off]; }
    __syncthreads();
  }
  float mean = rs_[0] * (1.0f/Eq);
  float var  = rq_[0] * (1.0f/Eq) - mean*mean;
  float rstd = rsqrtf(var + 1e-5f);
  #pragma unroll
  for (int j = 0; j < 4; ++j){
    int cc = tid + j*256;
    float o = x[(long long)row*Eq + cc] + (v[j]-mean)*rstd*g[cc] + b[cc];
    adn1[(long long)row*Eq + cc] = o;
    adn1b[(long long)row*Eq + cc] = f2bf(o);
  }
}

// out = adn1 + LN2( LNff(ffp)*gff+bff )*g2 + b2
__global__ __launch_bounds__(256) void ffnorm_kernel(
    const float* __restrict__ ffp, const float* __restrict__ adn1,
    const float* __restrict__ gff, const float* __restrict__ bff,
    const float* __restrict__ g2,  const float* __restrict__ b2,
    float* __restrict__ out){
  __shared__ float rs_[256], rq_[256];
  const int row = blockIdx.x, tid = threadIdx.x;
  const float* fr = ffp + (long long)row * Eq;
  float v[4]; float s = 0.f, sq = 0.f;
  #pragma unroll
  for (int j = 0; j < 4; ++j){ v[j] = fr[tid + j*256]; s += v[j]; sq += v[j]*v[j]; }
  rs_[tid] = s; rq_[tid] = sq; __syncthreads();
  for (int off = 128; off > 0; off >>= 1){
    if (tid < off){ rs_[tid] += rs_[tid+off]; rq_[tid] += rq_[tid+off]; }
    __syncthreads();
  }
  float mean = rs_[0] * (1.0f/Eq);
  float var  = rq_[0] * (1.0f/Eq) - mean*mean;
  float rstd = rsqrtf(var + 1e-5f);
  __syncthreads();
  float t[4]; s = 0.f; sq = 0.f;
  #pragma unroll
  for (int j = 0; j < 4; ++j){
    int cc = tid + j*256;
    t[j] = (v[j]-mean)*rstd*gff[cc] + bff[cc];
    s += t[j]; sq += t[j]*t[j];
  }
  rs_[tid] = s; rq_[tid] = sq; __syncthreads();
  for (int off = 128; off > 0; off >>= 1){
    if (tid < off){ rs_[tid] += rs_[tid+off]; rq_[tid] += rq_[tid+off]; }
    __syncthreads();
  }
  float mean2 = rs_[0] * (1.0f/Eq);
  float var2  = rq_[0] * (1.0f/Eq) - mean2*mean2;
  float rstd2 = rsqrtf(var2 + 1e-5f);
  #pragma unroll
  for (int j = 0; j < 4; ++j){
    int cc = tid + j*256;
    out[(long long)row*Eq + cc] = adn1[(long long)row*Eq + cc] + (t[j]-mean2)*rstd2*g2[cc] + b2[cc];
  }
}

extern "C" void kernel_launch(void* const* d_in, const int* in_sizes, int n_in,
                              void* d_out, int out_size, void* d_ws, size_t ws_size,
                              hipStream_t stream)
{
  (void)in_sizes; (void)n_in; (void)out_size; (void)ws_size;
  const float* x   = (const float*)d_in[0];
  const float* Wq  = (const float*)d_in[1];  const float* bq  = (const float*)d_in[2];
  const float* Wk  = (const float*)d_in[3];  const float* bk  = (const float*)d_in[4];
  const float* Wv  = (const float*)d_in[5];  const float* bv  = (const float*)d_in[6];
  const float* W1  = (const float*)d_in[7];  const float* b1  = (const float*)d_in[8];
  const float* W2  = (const float*)d_in[9];  const float* b2  = (const float*)d_in[10];
  const float* gff = (const float*)d_in[11]; const float* bff = (const float*)d_in[12];
  const float* g1  = (const float*)d_in[13]; const float* bb1 = (const float*)d_in[14];
  const float* g2  = (const float*)d_in[15]; const float* bb2 = (const float*)d_in[16];
  float* out = (float*)d_out;

  char* p = (char*)d_ws;
  auto carve = [&](size_t bytes) -> char* {
    char* r = p; p += (bytes + 255) & ~(size_t)255; return r;
  };
  unsigned short* xb    = (unsigned short*)carve((size_t)BSq*Eq*2);
  unsigned short* wqt   = (unsigned short*)carve((size_t)Eq*Eq*2);
  unsigned short* wkt   = (unsigned short*)carve((size_t)Eq*Eq*2);
  unsigned short* wvt   = (unsigned short*)carve((size_t)Eq*Eq*2);
  unsigned short* w1t   = (unsigned short*)carve((size_t)Eq*Fq*2);   // [F,E]
  unsigned short* w2t   = (unsigned short*)carve((size_t)Fq*Eq*2);   // [E,F]
  unsigned short* Qb    = (unsigned short*)carve((size_t)BSq*Eq*2);
  unsigned short* Kb    = (unsigned short*)carve((size_t)BSq*Eq*2);
  unsigned short* Vb    = (unsigned short*)carve((size_t)BSq*Eq*2);
  unsigned short* Vt    = (unsigned short*)carve((size_t)Bq*Hq*Dq*Sq*2); // [BH, D, S]
  unsigned short* Vs    = (unsigned short*)carve((size_t)Bq*Hq*Dq*Sq*2); // Vt / Z
  unsigned short* eb    = (unsigned short*)carve((size_t)Bq*Hq*Sq*Sq*2); // exp(scores), unnormalized
  unsigned short* adn1b = (unsigned short*)carve((size_t)BSq*Eq*2);
  unsigned short* hb    = (unsigned short*)carve((size_t)BSq*Fq*2);
  float* Zbuf   = (float*)carve((size_t)Bq*Hq*Sq*4);
  float* attout = (float*)carve((size_t)BSq*Eq*4);
  float* adn1   = (float*)carve((size_t)BSq*Eq*4);
  float* ffp    = (float*)carve((size_t)BSq*Eq*4);

  dim3 blk(256);

  // conversions (+ weight transposes so every GEMM B-operand is [N,K] row-major)
  cvt_bf16_kernel<<<2048, blk, 0, stream>>>(x, xb, BSq*Eq);
  tcvt_kernel<<<dim3(Eq/32, Eq/32), blk, 0, stream>>>(Wq, wqt, Eq, Eq);
  tcvt_kernel<<<dim3(Eq/32, Eq/32), blk, 0, stream>>>(Wk, wkt, Eq, Eq);
  tcvt_kernel<<<dim3(Eq/32, Eq/32), blk, 0, stream>>>(Wv, wvt, Eq, Eq);
  tcvt_kernel<<<dim3(Fq/32, Eq/32), blk, 0, stream>>>(W1, w1t, Eq, Fq);
  tcvt_kernel<<<dim3(Eq/32, Fq/32), blk, 0, stream>>>(W2, w2t, Fq, Eq);

  // Q/K/V projections: [8192,1024] = x @ W + b, bf16 out
  gemm_bf16_wmma<<<dim3(Eq/64, BSq/128, 1), blk, 0, stream>>>(
      xb, Eq, 0, 0, wqt, Eq, 0, 0, bq, Qb, Eq, 0, 0, 1, BSq, Eq, Eq, 1, 1.0f, 0);
  gemm_bf16_wmma<<<dim3(Eq/64, BSq/128, 1), blk, 0, stream>>>(
      xb, Eq, 0, 0, wkt, Eq, 0, 0, bk, Kb, Eq, 0, 0, 1, BSq, Eq, Eq, 1, 1.0f, 0);
  gemm_bf16_wmma<<<dim3(Eq/64, BSq/128, 1), blk, 0, stream>>>(
      xb, Eq, 0, 0, wvt, Eq, 0, 0, bv, Vb, Eq, 0, 0, 1, BSq, Eq, Eq, 1, 1.0f, 0);

  // V -> [BH, D, S] so attention B-operand is [N,K] too
  vtrans_kernel<<<dim3(Sq/32, Dq/32, Bq*Hq), blk, 0, stream>>>(Vb, Vt);

  // e[z,q,k] = exp((q_bh @ k_bh^T)/10)  -- bf16, fused exp epilogue, no fp32 scores buffer
  gemm_bf16_wmma<<<dim3(Sq/64, Sq/128, Bq*Hq), blk, 0, stream>>>(
      Qb, Eq, (long long)Sq*Eq, Dq,
      Kb, Eq, (long long)Sq*Eq, Dq,
      nullptr,
      eb, Sq, (long long)Hq*Sq*Sq, (long long)Sq*Sq, 1,
      Sq, Sq, Dq, Hq, 0.1f, 2);

  // softmax normalizer over the query axis: Z[z,k] = sum_q e[z,q,k]
  colsum_kernel<<<dim3(Sq/256, Bq*Hq), blk, 0, stream>>>(eb, Zbuf);

  // fold 1/Z into V (Z lives on the contraction axis of att@V)
  vscale_kernel<<<dim3(Dq, Bq*Hq), blk, 0, stream>>>(Vt, Zbuf, Vs);

  // att_out[b,q,h*64+d] = e @ (v/Z)
  gemm_bf16_wmma<<<dim3(Dq/64, Sq/128, Bq*Hq), blk, 0, stream>>>(
      eb, Sq, (long long)Hq*Sq*Sq, (long long)Sq*Sq,
      Vs, Sq, (long long)Hq*Dq*Sq, (long long)Dq*Sq,
      nullptr,
      attout, Eq, (long long)Sq*Eq, Dq, 0,
      Sq, Dq, Sq, Hq, 1.0f, 0);

  // adn1 = x + LN(att_out)
  addnorm1_kernel<<<BSq, blk, 0, stream>>>(x, attout, g1, bb1, adn1, adn1b);

  // FFN1: h = gelu(adn1 @ W1 + b1), bf16 out ; B-operand = w1t [F,E]
  gemm_bf16_wmma<<<dim3(Fq/64, BSq/128, 1), blk, 0, stream>>>(
      adn1b, Eq, 0, 0, w1t, Eq, 0, 0, b1, hb, Fq, 0, 0, 1, BSq, Fq, Eq, 1, 1.0f, 1);

  // FFN2: ffp = h @ W2 + b2, fp32 out ; B-operand = w2t [E,F]
  gemm_bf16_wmma<<<dim3(Eq/64, BSq/128, 1), blk, 0, stream>>>(
      hb, Fq, 0, 0, w2t, Fq, 0, 0, b2, ffp, Eq, 0, 0, 0, BSq, Eq, Fq, 1, 1.0f, 0);

  // out = adn1 + LN2(LNff(ffp))
  ffnorm_kernel<<<BSq, blk, 0, stream>>>(ffp, adn1, gff, bff, g2, bb2, out);
}
